// TextEncoder_70635032150365
// MI455X (gfx1250) — matrix operand order repaired and checked by reading, not compiled, not run on hardware
//
#include <hip/hip_runtime.h>
#include <cstdint>
#include <cstddef>

// ---------------- constants for this model ----------------
#define BATCH 8
#define SEQ   1024
#define DMODEL 768
#define NHEAD 12
#define HDIM  64
#define NLAYER 6
#define FFDIM 3072
#define ROWS (BATCH*SEQ)          // 8192
#define QKVW (3*DMODEL)           // 2304

typedef __attribute__((ext_vector_type(16))) __bf16 v16bf;
typedef __attribute__((ext_vector_type(8)))  float  v8f;

#define DEV __device__ __forceinline__

// CDNA5 WMMA bf16 fragment index maps (wave32, 16x16x32)
DEV int a_kidx(int lane, int j) { return j + ((j >= 8) ? 8 : 0) + ((lane >= 16) ? 8 : 0); }
DEV int b_kidx(int lane, int j) { return j + ((j >= 8) ? 8 : 0) + ((lane >= 16) ? 8 : 0); }

DEV v8f wmma_bf16(v16bf a, v16bf b, v8f c) {
  // v_wmma_f32_16x16x32_bf16 : D = A x B + C
  return __builtin_amdgcn_wmma_f32_16x16x32_bf16(false, a, false, b, (short)0, c,
                                                 false, false);
}

DEV float gelu_exact(float x) {
  return 0.5f * x * (1.0f + erff(x * 0.70710678118654752f));
}

// flat pointer -> wave-relative LDS byte offset (ISA 10.2: low 32 bits of the address)
DEV uint32_t lds_off(const void* p) { return (uint32_t)(uintptr_t)p; }

// CDNA5 async memory->LDS copy, 16 bytes per lane, tracked by ASYNCcnt
DEV void async_ld_b128(uint32_t lds_byte, uint32_t glb_byte_off, const void* base) {
  asm volatile("global_load_async_to_lds_b128 %0, %1, %2"
               :
               : "v"(lds_byte), "v"(glb_byte_off), "s"(base)
               : "memory");
}
DEV void wait_async0() { asm volatile("s_wait_asynccnt 0x0" ::: "memory"); }

// ---------------- embedding + positional encoding ----------------
__global__ __launch_bounds__(256)
void te_embed_kernel(const int* __restrict__ tokens, const float* __restrict__ emb,
                     const float* __restrict__ pe, float* __restrict__ x) {
  int row = blockIdx.x;                // 0..ROWS-1
  int s = row & (SEQ - 1);
  int t = tokens[row];
  const float* er = emb + (size_t)t * DMODEL;
  const float* pr = pe + (size_t)s * DMODEL;
  float* xr = x + (size_t)row * DMODEL;
  for (int d = threadIdx.x; d < DMODEL; d += 256) xr[d] = er[d] + pr[d];
}

// ---------------- layernorm (fp32 in -> bf16 out) ----------------
__global__ __launch_bounds__(256)
void te_ln_kernel(const float* __restrict__ x, const float* __restrict__ sc,
                  const float* __restrict__ bi, __bf16* __restrict__ out) {
  __shared__ float ssum[256];
  __shared__ float ssq[256];
  int row = blockIdx.x, tid = threadIdx.x;
  const float* xr = x + (size_t)row * DMODEL;
  float v0 = xr[tid], v1 = xr[tid + 256], v2 = xr[tid + 512];
  ssum[tid] = v0 + v1 + v2;
  ssq[tid]  = v0 * v0 + v1 * v1 + v2 * v2;
  __syncthreads();
  for (int st = 128; st > 0; st >>= 1) {
    if (tid < st) { ssum[tid] += ssum[tid + st]; ssq[tid] += ssq[tid + st]; }
    __syncthreads();
  }
  float mean = ssum[0] * (1.0f / DMODEL);
  float var  = ssq[0] * (1.0f / DMODEL) - mean * mean;
  float inv  = rsqrtf(var + 1e-5f);
  __bf16* orow = out + (size_t)row * DMODEL;
  orow[tid]       = (__bf16)((v0 - mean) * inv * sc[tid]       + bi[tid]);
  orow[tid + 256] = (__bf16)((v1 - mean) * inv * sc[tid + 256] + bi[tid + 256]);
  orow[tid + 512] = (__bf16)((v2 - mean) * inv * sc[tid + 512] + bi[tid + 512]);
}

// ---------------- weight conversion (fp32 -> bf16) ----------------
// wq/wk/wv [H,D,HD] (one layer) -> fused [D, 3*D] with cols [q|k|v], col = h*HD+e
__global__ __launch_bounds__(256)
void te_convert_qkvw_kernel(const float* __restrict__ wq, const float* __restrict__ wk,
                            const float* __restrict__ wv, __bf16* __restrict__ dst) {
  int i = blockIdx.x * 256 + threadIdx.x;
  if (i >= DMODEL * QKVW) return;
  int col = i % QKVW, d = i / QKVW;
  int which = col / DMODEL;
  int hh = (col % DMODEL) / HDIM;
  int e  = col % HDIM;
  const float* src = (which == 0) ? wq : ((which == 1) ? wk : wv);
  dst[i] = (__bf16)src[((size_t)hh * DMODEL + d) * HDIM + e];
}

__global__ __launch_bounds__(256)
void te_convert_kernel(const float* __restrict__ src, __bf16* __restrict__ dst, int n) {
  int i = blockIdx.x * 256 + threadIdx.x;
  if (i < n) dst[i] = (__bf16)src[i];
}

// ---------------- WMMA GEMM: C[M,N] = A[M,K] x W[K,N] ----------------
// Block tile 128x64, 4 waves; each wave owns 32x64 (2 A-frags x 4 B-frags,
// 8 wmma per K-step). Tiles staged memory->LDS with CDNA5 async b128 copies,
// double-buffered so the next K-tile streams in under the current WMMAs.
// EPI bit0: +bias[col]   bit1: gelu   bit2: accumulate into fp32 outF (residual)
// else: write bf16 outB
template <int EPI>
__global__ __launch_bounds__(128)
void te_gemm_kernel(const __bf16* __restrict__ A, const __bf16* __restrict__ W,
                    int M, int N, int K, const float* __restrict__ bias,
                    float* __restrict__ outF, __bf16* __restrict__ outB) {
  __shared__ __align__(16) __bf16 As[2][128][40];  // 128x32 tile, 80B row pitch (16B mult)
  __shared__ __align__(16) __bf16 Ws[2][32][72];   // 32x64 tile, 144B row pitch (16B mult)
  int tid = threadIdx.x, lane = tid & 31, w = tid >> 5;
  int brow = blockIdx.x * 128, bcol = blockIdx.y * 64;

  // issue async copies for K-tile starting at k0 into buffer `buf`
  auto issue = [&](int k0, int buf) {
    // A tile: 128 rows x 4 x (8 bf16 = 16B) segments = 512 lane-transfers
    for (int i = tid; i < 512; i += 128) {
      int r = i >> 2, seg = i & 3;
      uint32_t l = lds_off(&As[buf][r][seg * 8]);
      uint32_t g = (uint32_t)((((size_t)(brow + r)) * K + k0 + seg * 8) * 2);
      async_ld_b128(l, g, (const void*)A);
    }
    // W tile: 32 rows x 8 segments = 256 lane-transfers
    for (int i = tid; i < 256; i += 128) {
      int r = i >> 3, seg = i & 7;
      uint32_t l = lds_off(&Ws[buf][r][seg * 8]);
      uint32_t g = (uint32_t)((((size_t)(k0 + r)) * N + bcol + seg * 8) * 2);
      async_ld_b128(l, g, (const void*)W);
    }
  };

  v8f acc[2][4] = {};
  issue(0, 0);
  wait_async0();
  __syncthreads();

  int ib = 0;
  for (int k0 = 0; k0 < K; k0 += 32, ib ^= 1) {
    if (k0 + 32 < K) issue(k0 + 32, ib ^ 1);   // prefetch next tile (async)
    v16bf a[2];
#pragma unroll
    for (int mt = 0; mt < 2; ++mt)
#pragma unroll
      for (int j = 0; j < 16; ++j)
        a[mt][j] = As[ib][w * 32 + mt * 16 + (lane & 15)][a_kidx(lane, j)];
#pragma unroll
    for (int nt = 0; nt < 4; ++nt) {
      v16bf bm;
#pragma unroll
      for (int j = 0; j < 16; ++j) bm[j] = Ws[ib][b_kidx(lane, j)][nt * 16 + (lane & 15)];
      acc[0][nt] = wmma_bf16(a[0], bm, acc[0][nt]);
      acc[1][nt] = wmma_bf16(a[1], bm, acc[1][nt]);
    }
    wait_async0();      // this wave's prefetch done
    __syncthreads();    // everyone done with current + prefetch buffers
  }

#pragma unroll
  for (int mt = 0; mt < 2; ++mt) {
#pragma unroll
    for (int nt = 0; nt < 4; ++nt) {
#pragma unroll
      for (int r = 0; r < 8; ++r) {
        int m = w * 32 + mt * 16 + r + ((lane >= 16) ? 8 : 0);
        int n = nt * 16 + (lane & 15);
        size_t row = (size_t)(brow + m), col = (size_t)(bcol + n);
        float v = acc[mt][nt][r];
        if constexpr ((EPI & 1) != 0) v += bias[col];
        if constexpr ((EPI & 2) != 0) v = gelu_exact(v);
        if constexpr ((EPI & 4) != 0) outF[row * N + col] += v;
        else                          outB[row * N + col] = (__bf16)v;
      }
    }
  }
}

// ---------------- attention: full score rows resident in LDS ----------------
// grid: x = query block (S/64), y = b*NHEAD + h ; block = 128 threads (4 waves)
// dyn LDS = 64 * SEQ * 4 = 256 KB (of the 320 KB WGP budget)
__global__ __launch_bounds__(128)
void te_attn_kernel(const __bf16* __restrict__ qkv, const int* __restrict__ kmask,
                    __bf16* __restrict__ o) {
  extern __shared__ float sc[];    // [64][SEQ]
  int tid = threadIdx.x, lane = tid & 31, w = tid >> 5;
  int qb = blockIdx.x;
  int bh = blockIdx.y;
  int b = bh / NHEAD, h = bh % NHEAD;
  size_t base = (size_t)b * SEQ * QKVW;
  int qrow0 = qb * 64;

  // ---- pass A: scores = Q K^T / sqrt(HD), masked ----
  for (int kt = 0; kt < SEQ / 16; ++kt) {
    v8f acc = {};
#pragma unroll
    for (int ks = 0; ks < 2; ++ks) {       // HD=64 -> 2 K-steps of 32
      v16bf a, bm;
      int qr = qrow0 + w * 16 + (lane & 15);
#pragma unroll
      for (int j = 0; j < 16; ++j)
        a[j] = qkv[base + (size_t)qr * QKVW + h * HDIM + ks * 32 + a_kidx(lane, j)];
      int key = kt * 16 + (lane & 15);
#pragma unroll
      for (int j = 0; j < 16; ++j)
        bm[j] = qkv[base + (size_t)key * QKVW + DMODEL + h * HDIM + ks * 32 + b_kidx(lane, j)];
      acc = wmma_bf16(a, bm, acc);
    }
#pragma unroll
    for (int r = 0; r < 8; ++r) {
      int m = w * 16 + r + ((lane >= 16) ? 8 : 0);
      int key = kt * 16 + (lane & 15);
      float s = acc[r] * 0.125f;
      if (kmask[b * SEQ + key] == 0) s = -3.0e38f;
      sc[(size_t)m * SEQ + key] = s;
    }
  }
  __syncthreads();

  // ---- softmax over each of the 64 rows ----
  if (tid < 64) {
    float* row = &sc[(size_t)tid * SEQ];
    float mx = -3.4e38f;
    for (int i = 0; i < SEQ; ++i) mx = fmaxf(mx, row[i]);
    float sum = 0.0f;
    for (int i = 0; i < SEQ; ++i) { float e = __expf(row[i] - mx); row[i] = e; sum += e; }
    float inv = 1.0f / sum;
    for (int i = 0; i < SEQ; ++i) row[i] *= inv;
  }
  __syncthreads();

  // ---- pass C: O = P V ----
  v8f acc[4] = {};
  for (int kt = 0; kt < SEQ / 32; ++kt) {
    v16bf a;
#pragma unroll
    for (int j = 0; j < 16; ++j)
      a[j] = (__bf16)sc[(size_t)(w * 16 + (lane & 15)) * SEQ + kt * 32 + a_kidx(lane, j)];
#pragma unroll
    for (int nt = 0; nt < 4; ++nt) {
      v16bf bm;
#pragma unroll
      for (int j = 0; j < 16; ++j) {
        int key = kt * 32 + b_kidx(lane, j);
        bm[j] = qkv[base + (size_t)key * QKVW + 2 * DMODEL + h * HDIM + nt * 16 + (lane & 15)];
      }
      acc[nt] = wmma_bf16(a, bm, acc[nt]);
    }
  }
#pragma unroll
  for (int nt = 0; nt < 4; ++nt) {
#pragma unroll
    for (int r = 0; r < 8; ++r) {
      int srow = qrow0 + w * 16 + r + ((lane >= 16) ? 8 : 0);
      int col = h * HDIM + nt * 16 + (lane & 15);
      o[((size_t)b * SEQ + srow) * DMODEL + col] = (__bf16)acc[nt][r];
    }
  }
}

// ---------------- host side ----------------
extern "C" void kernel_launch(void* const* d_in, const int* in_sizes, int n_in,
                              void* d_out, int out_size, void* d_ws, size_t ws_size,
                              hipStream_t stream) {
  const int*   tokens = (const int*)d_in[0];
  const int*   kmask  = (const int*)d_in[1];
  const float* emb    = (const float*)d_in[2];
  const float* pe     = (const float*)d_in[3];
  const float* ln1_s  = (const float*)d_in[4];
  const float* ln1_b  = (const float*)d_in[5];
  const float* wq     = (const float*)d_in[6];
  const float* wk     = (const float*)d_in[7];
  const float* wv     = (const float*)d_in[8];
  const float* wo     = (const float*)d_in[9];
  const float* ln2_s  = (const float*)d_in[10];
  const float* ln2_b  = (const float*)d_in[11];
  const float* w1     = (const float*)d_in[12];
  const float* b1     = (const float*)d_in[13];
  const float* w2     = (const float*)d_in[14];
  const float* b2     = (const float*)d_in[15];

  float* x = (float*)d_out;                 // residual stream [ROWS, DMODEL]

  // workspace layout (bf16 buffers)
  char* ws = (char*)d_ws;
  size_t off = 0;
  auto alloc = [&](size_t elems) { __bf16* p = (__bf16*)(ws + off); off += elems * 2; return p; };
  __bf16* xn   = alloc((size_t)ROWS * DMODEL);
  __bf16* qkvb = alloc((size_t)ROWS * QKVW);
  __bf16* ob   = alloc((size_t)ROWS * DMODEL);
  __bf16* hb   = alloc((size_t)ROWS * FFDIM);
  __bf16* Wqkv = alloc((size_t)DMODEL * QKVW);
  __bf16* Wo   = alloc((size_t)DMODEL * DMODEL);
  __bf16* W1b  = alloc((size_t)DMODEL * FFDIM);
  __bf16* W2b  = alloc((size_t)FFDIM * DMODEL);
  (void)ws_size;

  te_embed_kernel<<<ROWS, 256, 0, stream>>>(tokens, emb, pe, x);

  const size_t projSz = (size_t)NHEAD * DMODEL * HDIM;   // per-layer wq/wk/wv elems
  for (int l = 0; l < NLAYER; ++l) {
    // weight conversion for this layer
    {
      int n = DMODEL * QKVW;
      te_convert_qkvw_kernel<<<(n + 255) / 256, 256, 0, stream>>>(
          wq + (size_t)l * projSz, wk + (size_t)l * projSz, wv + (size_t)l * projSz, Wqkv);
    }
    { int n = DMODEL * DMODEL;
      te_convert_kernel<<<(n + 255) / 256, 256, 0, stream>>>(wo + (size_t)l * DMODEL * DMODEL, Wo, n); }
    { int n = DMODEL * FFDIM;
      te_convert_kernel<<<(n + 255) / 256, 256, 0, stream>>>(w1 + (size_t)l * DMODEL * FFDIM, W1b, n); }
    { int n = FFDIM * DMODEL;
      te_convert_kernel<<<(n + 255) / 256, 256, 0, stream>>>(w2 + (size_t)l * FFDIM * DMODEL, W2b, n); }

    // ln1 -> xn (bf16)
    te_ln_kernel<<<ROWS, 256, 0, stream>>>(x, ln1_s + (size_t)l * DMODEL,
                                           ln1_b + (size_t)l * DMODEL, xn);
    // fused QKV projection: [8192,768] x [768,2304]
    te_gemm_kernel<0><<<dim3(ROWS / 128, QKVW / 64), 128, 0, stream>>>(
        xn, Wqkv, ROWS, QKVW, DMODEL, nullptr, nullptr, qkvb);
    // attention (scores resident in 256KB LDS)
    te_attn_kernel<<<dim3(SEQ / 64, BATCH * NHEAD), 128, 64 * SEQ * sizeof(float), stream>>>(
        qkvb, kmask, ob);
    // output projection, residual accumulate: x += o @ Wo
    te_gemm_kernel<4><<<dim3(ROWS / 128, DMODEL / 64), 128, 0, stream>>>(
        ob, Wo, ROWS, DMODEL, DMODEL, nullptr, x, nullptr);
    // ln2 -> xn
    te_ln_kernel<<<ROWS, 256, 0, stream>>>(x, ln2_s + (size_t)l * DMODEL,
                                           ln2_b + (size_t)l * DMODEL, xn);
    // FFN1 + bias + exact GELU -> h (bf16)
    te_gemm_kernel<3><<<dim3(ROWS / 128, FFDIM / 64), 128, 0, stream>>>(
        xn, W1b, ROWS, FFDIM, DMODEL, b1 + (size_t)l * FFDIM, nullptr, hb);
    // FFN2 + bias, residual accumulate: x += h @ W2 + b2
    te_gemm_kernel<5><<<dim3(ROWS / 128, DMODEL / 64), 128, 0, stream>>>(
        hb, W2b, ROWS, DMODEL, FFDIM, b2 + (size_t)l * DMODEL, x, nullptr);
  }
}
